// TurboQuantKVCache_59734405153118
// MI455X (gfx1250) — matrix operand order repaired and checked by reading, not compiled, not run on hardware
//
#include <hip/hip_runtime.h>
#include <hip/hip_bf16.h>

typedef __attribute__((ext_vector_type(16))) _Float16 v16h;
typedef __attribute__((ext_vector_type(8)))  float    v8f;

#define TQ_D  128   // head dim
#define TQ_PD 64    // packed int32 words per token

// ---------------------------------------------------------------------------
// inverse scatter map: inv[pos] = new-token index, or -1
// ---------------------------------------------------------------------------
__global__ void __launch_bounds__(256)
tq_inv_init(int* __restrict__ inv, int S_max) {
    int i = blockIdx.x * 256 + threadIdx.x;
    if (i < S_max) inv[i] = -1;
}

__global__ void __launch_bounds__(256)
tq_inv_scatter(const int* __restrict__ input_pos, int* __restrict__ inv,
               int S_new, int S_max) {
    int i = blockIdx.x * 256 + threadIdx.x;
    if (i < S_new) {
        int p = input_pos[i];
        if (p >= 0 && p < S_max) inv[p] = i;
    }
}

// ---------------------------------------------------------------------------
// Quantize: per wave, 16 tokens.  x_rot = x_norm @ rotation via WMMA f16.
// ---------------------------------------------------------------------------
__global__ void __launch_bounds__(256)
tq_quant(const float* __restrict__ k_val, const float* __restrict__ v_val,
         const float* __restrict__ rotation, const float* __restrict__ boundaries,
         int* __restrict__ kp_new, int* __restrict__ vp_new,
         float* __restrict__ kmag_new, float* __restrict__ vmag_new,
         float* __restrict__ kmean_new, float* __restrict__ vmean_new,
         int BH, int S_new)
{
    __shared__ __attribute__((aligned(32))) _Float16 ldsB[TQ_D * TQ_D]; // 32 KB
    for (int i = threadIdx.x; i < TQ_D * TQ_D; i += 256) {
        int n = i >> 7, k = i & 127;
        ldsB[i] = (_Float16)rotation[k * TQ_D + n];   // B[k][n] = rot[k][n]
    }
    __syncthreads();

    float bnd[15];
#pragma unroll
    for (int j = 0; j < 15; ++j) bnd[j] = boundaries[j];

    const int lane = threadIdx.x & 31;
    const int hi   = lane >> 4;
    const int nl   = lane & 15;
    const int tilesPerSeq    = S_new >> 4;
    const int tilesPerTensor = BH * tilesPerSeq;

    int w = blockIdx.x * 8 + (threadIdx.x >> 5);
    if (w >= 2 * tilesPerTensor) return;
    int tensor = w / tilesPerTensor;
    int tid    = w - tensor * tilesPerTensor;
    int bh     = tid / tilesPerSeq;
    int st     = tid - bh * tilesPerSeq;

    long tokBase = (long)bh * S_new + (long)st * 16;
    const float* x     = (tensor ? v_val : k_val) + tokBase * TQ_D;
    int*         p_out = (tensor ? vp_new : kp_new) + tokBase * TQ_PD;
    float*       magO  = (tensor ? vmag_new : kmag_new) + tokBase;
    float*       meanO = (tensor ? vmean_new : kmean_new) + tokBase;

    // warm the tile into cache (global_prefetch_b8): 16 rows x 512 B
    __builtin_prefetch(x + lane * (2048 / 32), 0, 3);

    // per-token mean / sigma via wave32 butterfly
    float mean_own = 0.f, scale_own = 0.f;
#pragma unroll 1
    for (int m = 0; m < 16; ++m) {
        float s = 0.f, sq = 0.f;
#pragma unroll
        for (int j = 0; j < 4; ++j) {
            float v = x[m * TQ_D + lane + 32 * j];
            s += v; sq += v * v;
        }
#pragma unroll
        for (int off = 16; off >= 1; off >>= 1) {
            s  += __shfl_xor(s,  off, 32);
            sq += __shfl_xor(sq, off, 32);
        }
        float mean = s * (1.0f / 128.0f);
        float var  = sq - 128.0f * mean * mean;
        float mag  = fmaxf(sqrtf(fmaxf(var, 0.0f)), 1e-8f);
        if (lane == 0) { magO[m] = mag; meanO[m] = mean; }
        if (m == nl) { mean_own = mean; scale_own = 11.313708498984760f / mag; }
    }

    // A fragments: 16x32 f16, contiguous K (lanes 0-15: K 0..15; 16-31: K 16..31)
    v16h a[4];
    {
        const float* xr = x + nl * TQ_D;
#pragma unroll
        for (int kt = 0; kt < 4; ++kt) {
            int kb = kt * 32 + hi * 16;
#pragma unroll
            for (int i = 0; i < 16; ++i) {
                int k = kb + i;
                a[kt][i] = (_Float16)((xr[k] - mean_own) * scale_own);
            }
        }
    }

#pragma unroll 2
    for (int nt = 0; nt < 8; ++nt) {
        // preload all 4 B fragments so the 8 ds_load_b128 batch ahead of WMMA
        v16h bf[4];
#pragma unroll
        for (int kt = 0; kt < 4; ++kt) {
            const _Float16* bp = &ldsB[(nt * 16 + nl) * TQ_D + kt * 32 + hi * 16];
            bf[kt] = *(const v16h*)bp;
        }
        v8f acc = {};
#pragma unroll
        for (int kt = 0; kt < 4; ++kt)
            acc = __builtin_amdgcn_wmma_f32_16x16x32_f16(false, a[kt], false, bf[kt],
                                                         (short)0, acc, false, false);
        // bucketize (searchsorted left == count of boundaries < v) and pack nibbles
#pragma unroll
        for (int r = 0; r < 8; ++r) {
            float val = acc[r];
            int idx = 0;
#pragma unroll
            for (int j = 0; j < 15; ++j) idx += (bnd[j] < val) ? 1 : 0;
            int idx_hi = __shfl_xor(idx, 1, 32);
            if ((nl & 1) == 0) {
                int m = r + hi * 8;
                int n = nt * 16 + nl;
                p_out[m * TQ_PD + (n >> 1)] = idx | (idx_hi << 4);
            }
        }
    }
}

// ---------------------------------------------------------------------------
// Dequantize whole cache via WMMA against rotation^T in LDS.
// ---------------------------------------------------------------------------
__global__ void __launch_bounds__(256)
tq_dequant(const int* __restrict__ kp_in, const int* __restrict__ vp_in,
           const float* __restrict__ kmag_in, const float* __restrict__ vmag_in,
           const float* __restrict__ kmean_in, const float* __restrict__ vmean_in,
           const int* __restrict__ kp_new, const int* __restrict__ vp_new,
           const float* __restrict__ kmag_new, const float* __restrict__ vmag_new,
           const float* __restrict__ kmean_new, const float* __restrict__ vmean_new,
           const int* __restrict__ inv_map,
           const float* __restrict__ rotation, const float* __restrict__ centroids,
           float* __restrict__ out, int BH, int S_new, int S_max)
{
    __shared__ __attribute__((aligned(32))) _Float16 ldsB[TQ_D * TQ_D];
    __shared__ float ldsC[16];
    for (int i = threadIdx.x; i < TQ_D * TQ_D; i += 256)
        ldsB[i] = (_Float16)rotation[i];                    // B[k][n] = rot[n][k]
    if (threadIdx.x < 16) ldsC[threadIdx.x] = centroids[threadIdx.x];
    __syncthreads();

    const int lane = threadIdx.x & 31;
    const int hi   = lane >> 4;
    const int nl   = lane & 15;
    const int tilesPerSeq    = S_max >> 4;
    const int tilesPerTensor = BH * tilesPerSeq;

    int w = blockIdx.x * 8 + (threadIdx.x >> 5);
    if (w >= 2 * tilesPerTensor) return;
    int tensor = w / tilesPerTensor;
    int tid    = w - tensor * tilesPerTensor;
    int bh     = tid / tilesPerSeq;
    int pt     = tid - bh * tilesPerSeq;

    int pos = pt * 16 + nl;
    int j   = inv_map[pos];
    const int* psrc;
    float mag_own, mean_own;
    if (j >= 0) {
        long o = (long)bh * S_new + j;
        psrc     = (tensor ? vp_new : kp_new) + o * TQ_PD;
        mag_own  = (tensor ? vmag_new  : kmag_new)[o];
        mean_own = (tensor ? vmean_new : kmean_new)[o];
    } else {
        long o = (long)bh * S_max + pos;
        psrc     = (tensor ? vp_in : kp_in) + o * TQ_PD;
        mag_own  = (tensor ? vmag_in  : kmag_in)[o];
        mean_own = (tensor ? vmean_in : kmean_in)[o];
    }
    __builtin_prefetch(psrc, 0, 3);   // global_prefetch_b8: packed row (256 B)

    // A fragments: unpack nibble pairs, centroid lookup via LDS LUT
    v16h a[4];
#pragma unroll
    for (int kt = 0; kt < 4; ++kt) {
        int kb = kt * 32 + hi * 16;
#pragma unroll
        for (int ii = 0; ii < 8; ++ii) {
            int k0   = kb + 2 * ii;
            int word = psrc[k0 >> 1];
            a[kt][2 * ii]     = (_Float16)ldsC[word & 15];
            a[kt][2 * ii + 1] = (_Float16)ldsC[(word >> 4) & 15];
        }
    }

    // broadcast mag/mean of the 8 tokens this half-wave accumulates
    float magm[8], meanm[8];
#pragma unroll
    for (int r = 0; r < 8; ++r) {
        int src  = r + hi * 8;
        magm[r]  = __shfl(mag_own,  src, 32);
        meanm[r] = __shfl(mean_own, src, 32);
    }

    float* outp = out + (long)tensor * BH * S_max * TQ_D
                      + ((long)bh * S_max + (long)pt * 16) * TQ_D;
    const float inv_s = 0.08838834764831845f;   // 1/sqrt(128)

#pragma unroll 2
    for (int nt = 0; nt < 8; ++nt) {
        v16h bf[4];
#pragma unroll
        for (int kt = 0; kt < 4; ++kt) {
            const _Float16* bp = &ldsB[(nt * 16 + nl) * TQ_D + kt * 32 + hi * 16];
            bf[kt] = *(const v16h*)bp;
        }
        v8f acc = {};
#pragma unroll
        for (int kt = 0; kt < 4; ++kt)
            acc = __builtin_amdgcn_wmma_f32_16x16x32_f16(false, a[kt], false, bf[kt],
                                                         (short)0, acc, false, false);
#pragma unroll
        for (int r = 0; r < 8; ++r) {
            int m = r + hi * 8;
            int n = nt * 16 + nl;
            outp[m * TQ_D + n] = acc[r] * magm[r] * inv_s + meanm[r];
        }
    }
}

// ---------------------------------------------------------------------------
extern "C" void kernel_launch(void* const* d_in, const int* in_sizes, int n_in,
                              void* d_out, int out_size, void* d_ws, size_t ws_size,
                              hipStream_t stream)
{
    const int*   input_pos  = (const int*)  d_in[0];
    const float* k_val      = (const float*)d_in[1];
    const float* v_val      = (const float*)d_in[2];
    const float* rotation   = (const float*)d_in[3];
    const float* centroids  = (const float*)d_in[4];
    const float* boundaries = (const float*)d_in[5];
    const int*   k_packed   = (const int*)  d_in[6];
    const int*   v_packed   = (const int*)  d_in[7];
    const float* k_mag      = (const float*)d_in[8];
    const float* v_mag      = (const float*)d_in[9];
    const float* k_mean     = (const float*)d_in[10];
    const float* v_mean     = (const float*)d_in[11];

    const int S_new = in_sizes[0];
    const int BH    = (int)((long)in_sizes[1] / ((long)S_new * TQ_D));
    const int S_max = (int)((long)in_sizes[8] / BH);

    char* ws = (char*)d_ws;
    size_t off = 0;
    auto grab = [&](size_t bytes) {
        char* p = ws + off;
        off = (off + bytes + 255) & ~(size_t)255;
        return p;
    };
    int*   inv       = (int*)  grab((size_t)S_max * 4);
    int*   kp_new    = (int*)  grab((size_t)BH * S_new * TQ_PD * 4);
    int*   vp_new    = (int*)  grab((size_t)BH * S_new * TQ_PD * 4);
    float* kmag_new  = (float*)grab((size_t)BH * S_new * 4);
    float* vmag_new  = (float*)grab((size_t)BH * S_new * 4);
    float* kmean_new = (float*)grab((size_t)BH * S_new * 4);
    float* vmean_new = (float*)grab((size_t)BH * S_new * 4);
    (void)n_in; (void)out_size; (void)ws_size;

    tq_inv_init<<<(S_max + 255) / 256, 256, 0, stream>>>(inv, S_max);
    tq_inv_scatter<<<(S_new + 255) / 256, 256, 0, stream>>>(input_pos, inv, S_new, S_max);

    int qWaves = 2 * BH * (S_new / 16);
    tq_quant<<<(qWaves + 7) / 8, 256, 0, stream>>>(
        k_val, v_val, rotation, boundaries,
        kp_new, vp_new, kmag_new, vmag_new, kmean_new, vmean_new, BH, S_new);

    int dWaves = 2 * BH * (S_max / 16);
    tq_dequant<<<(dWaves + 7) / 8, 256, 0, stream>>>(
        k_packed, v_packed, k_mag, v_mag, k_mean, v_mean,
        kp_new, vp_new, kmag_new, vmag_new, kmean_new, vmean_new,
        inv, rotation, centroids, (float*)d_out, BH, S_new, S_max);
}
